// LSTMModelWithPolyBias_59296318488680
// MI455X (gfx1250) — matrix operand order repaired and checked by reading, not compiled, not run on hardware
//
#include <hip/hip_runtime.h>
#include <math.h>
#include <stdint.h>

typedef __attribute__((ext_vector_type(16))) _Float16 v16h;
typedef __attribute__((ext_vector_type(8)))  _Float16 v8h;
typedef __attribute__((ext_vector_type(8)))  float    v8f;

#define BM 128
#define BN 128
#define BK 32
#define SAS 40   // padded LDS row stride in halfs (80B) to dodge bank conflicts

// ------------------------------------------------- padded f32 -> f16 staging
// d[r][c] (drows x dcols, row-major) = (r<srows && c<scols) ? s[r][c] : 0
__global__ __launch_bounds__(256)
void cvt_pad_f16(const float* __restrict__ s, _Float16* __restrict__ d,
                 int srows, int scols, int drows, int dcols)
{
  int i = blockIdx.x * 256 + threadIdx.x;
  const int stride = gridDim.x * 256;
  const int n = drows * dcols;
  for (; i < n; i += stride) {
    const int r = i / dcols;
    const int c = i - r * dcols;
    d[i] = (r < srows && c < scols) ? (_Float16)s[(size_t)r * scols + c]
                                    : (_Float16)0.0f;
  }
}

// ------------------------------------------- async global->LDS (CDNA5, b128)
// Copies 32 bytes per lane: INST_OFFSET is added to both LDS and global
// addresses (ISA 15.18.3 / 4.4), so one base covers both halves.
__device__ __forceinline__ void async_copy_32B(uint32_t lds_addr,
                                               const _Float16* gptr)
{
  asm volatile(
      "global_load_async_to_lds_b128 %0, %1, off\n\t"
      "global_load_async_to_lds_b128 %0, %1, off offset:16"
      :
      : "v"(lds_addr), "v"((unsigned long long)(uintptr_t)gptr)
      : "memory");
}
__device__ __forceinline__ void wait_async0()
{
  asm volatile("s_wait_asynccnt 0x0" ::: "memory");
}

// ---------------------------------------------------------------- WMMA GEMM
// C[M,N] = A[M,Kp](f16,lda) @ B[Np,Kp](f16,ldb)^T + bias0 + bias1 + addM
// Preconditions (guaranteed by call sites):
//   grid.y*128 == M, Kp % 32 == 0, B has >= gridDim.x*128 rows (zero-padded),
//   all f16 rows 16B-aligned (lda/ldb multiples of 8 halfs).
// Hot loop: 4x global_load_async_to_lds_b128 (next tile, ASYNCcnt),
// 12x ds_load_b128, 8x v_wmma_f32_16x16x32_f16, s_wait_asynccnt, 1 barrier.
__global__ __launch_bounds__(256)
void wmma_gemm(const _Float16* __restrict__ A, int lda,
               const _Float16* __restrict__ Bm, int ldb,
               float* __restrict__ C, int ldc,
               const float* __restrict__ bias0,
               const float* __restrict__ bias1,
               const float* __restrict__ addM, int ldadd,
               int N, int Kp)
{
  __shared__ _Float16 sA[2][BM * SAS];
  __shared__ _Float16 sB[2][BN * SAS];

  const int tid  = threadIdx.x;
  const int wave = tid >> 5;
  const int lane = tid & 31;
  const int half = lane >> 4;     // 0/1: which 16-lane half
  const int l16  = lane & 15;
  const int wm   = wave >> 1;     // 0..3  (wave M tile)
  const int wn   = wave & 1;      // 0..1  (wave N tile)
  const int bM   = blockIdx.y * BM;
  const int bN   = blockIdx.x * BN;

  // loader mapping: 2 threads per row, 16 contiguous halfs (32B) each
  const int lr = tid >> 1;
  const int lc = (tid & 1) * 16;

  const _Float16* pa_cur = A  + (size_t)(bM + lr) * lda + lc;
  const _Float16* pb_cur = Bm + (size_t)(bN + lr) * ldb + lc;

  // per-lane LDS destination byte addresses for both buffers
  uint32_t ldsA[2], ldsB[2];
  ldsA[0] = (uint32_t)(uintptr_t)&sA[0][lr * SAS + lc];
  ldsA[1] = (uint32_t)(uintptr_t)&sA[1][lr * SAS + lc];
  ldsB[0] = (uint32_t)(uintptr_t)&sB[0][lr * SAS + lc];
  ldsB[1] = (uint32_t)(uintptr_t)&sB[1][lr * SAS + lc];

  v8f acc[2][4];
  {
    v8f z = {};
    #pragma unroll
    for (int i = 0; i < 2; ++i)
      #pragma unroll
      for (int j = 0; j < 4; ++j) acc[i][j] = z;
  }

  // ---- prologue: async-load tile 0 into buffer 0
  async_copy_32B(ldsA[0], pa_cur);
  async_copy_32B(ldsB[0], pb_cur);
  pa_cur += BK;
  pb_cur += BK;
  wait_async0();
  __syncthreads();

  const int nK = Kp / BK;
  for (int kb = 0; kb < nK; ++kb) {
    const int cur = kb & 1;

    // ---- issue next tile's async copies into the other buffer
    // (safe: that buffer was last read in iteration kb-1, fenced by barrier)
    if (kb + 1 < nK) {
      async_copy_32B(ldsA[cur ^ 1], pa_cur);
      async_copy_32B(ldsB[cur ^ 1], pb_cur);
      pa_cur += BK;
      pb_cur += BK;
    }

    // ---- fragments per CDNA5 16-bit WMMA VGPR layouts
    union V16 { v16h v; v8h h[2]; };
    V16 af[2], bf[4];
    #pragma unroll
    for (int i = 0; i < 2; ++i) {
      // A 16x32: lane<16 -> M=l16, K 0..7 & 16..23; lane>=16 -> K 8..15 & 24..31
      const _Float16* pa = &sA[cur][(wm * 32 + i * 16 + l16) * SAS + half * 8];
      af[i].h[0] = *(const v8h*)pa;
      af[i].h[1] = *(const v8h*)(pa + 16);
    }
    #pragma unroll
    for (int j = 0; j < 4; ++j) {
      // B 32x16: lane<16 -> N=l16, K 0..7 & 16..23; lane>=16 -> K 8..15 & 24..31
      const _Float16* pb = &sB[cur][(wn * 64 + j * 16 + l16) * SAS + half * 8];
      bf[j].h[0] = *(const v8h*)pb;
      bf[j].h[1] = *(const v8h*)(pb + 16);
    }

    #pragma unroll
    for (int i = 0; i < 2; ++i)
      #pragma unroll
      for (int j = 0; j < 4; ++j)
        acc[i][j] = __builtin_amdgcn_wmma_f32_16x16x32_f16(
            false, af[i].v, false, bf[j].v, (short)0, acc[i][j], false, false);

    // own async copies done, then cross-wave fence
    wait_async0();
    __syncthreads();
  }

  // ---- epilogue: C/D layout: VGPR v, lanes 0-15 -> M=v, lanes 16-31 -> M=v+8
  #pragma unroll
  for (int j = 0; j < 4; ++j) {
    const int gc = bN + wn * 64 + j * 16 + l16;
    if (gc >= N) continue;
    float badd = 0.0f;
    if (bias0) badd += bias0[gc];
    if (bias1) badd += bias1[gc];
    #pragma unroll
    for (int i = 0; i < 2; ++i) {
      const int gr0 = bM + wm * 32 + i * 16 + half * 8;
      #pragma unroll
      for (int v = 0; v < 8; ++v) {
        const int gr = gr0 + v;
        float val = acc[i][j][v] + badd;
        if (addM) val += addM[(size_t)gr * ldadd + gc];
        C[(size_t)gr * ldc + gc] = val;
      }
    }
  }
}

// ---------------------------------------------------------------- LSTM gates
__device__ __forceinline__ float sigf(float x) { return 1.0f / (1.0f + __expf(-x)); }

// gates: per-batch row stride grs, layout [i|f|g|o] each 300 wide (unpadded).
// c: [512*300] state (in/out). h16: f16 hidden at h16[b*hstride + j],
// j in [0,320): pad columns 300..319 are zero-filled (consumed as K by GEMMs).
__global__ __launch_bounds__(256)
void lstm_gates(const float* __restrict__ gates, int grs,
                float* __restrict__ c,
                _Float16* __restrict__ h16, int hstride,
                int first)
{
  const int idx = blockIdx.x * 256 + threadIdx.x;
  if (idx >= 512 * 320) return;
  const int b = idx / 320;
  const int j = idx - b * 320;
  if (j >= 300) {                       // zero the K padding
    h16[(size_t)b * hstride + j] = (_Float16)0.0f;
    return;
  }
  const float* g = gates + (size_t)b * grs;
  const float si = sigf(g[j]);
  const float sf = sigf(g[300 + j]);
  const float tg = tanhf(g[600 + j]);
  const float so = sigf(g[900 + j]);
  const int cidx = b * 300 + j;
  const float cp = first ? 0.0f : c[cidx];
  const float cn = sf * cp + si * tg;
  c[cidx] = cn;
  const float h = so * tanhf(cn);
  h16[(size_t)b * hstride + j] = (_Float16)h;
}

// ------------------------------------------------------- poly bias + W_b fix
__global__ __launch_bounds__(256)
void final_poly(const float* __restrict__ x, const float* __restrict__ fcout,
                float* __restrict__ out)
{
  __shared__ float rw[256];
  __shared__ float rp[256];
  const int b = blockIdx.x;
  const int tid = threadIdx.x;
  const float* xb = x + (size_t)b * 16 * 2500;
  const float* fb = fcout + (size_t)b * 2500;
  float wsum = 0.0f, psum = 0.0f;
  for (int d = tid; d < 2500; d += 256) {
    const float x13 = xb[13 * 2500 + d];
    const float x14 = xb[14 * 2500 + d];
    const float x15 = xb[15 * 2500 + d];
    const float poly = (x13 + x14 + x15) * (1.0f / 3.0f) + (x15 - x13);
    const float outb = fb[d] + poly;
    wsum += 1.0f - outb * outb;
    psum += xb[d] - outb;            // x[b,0,d] - outb
  }
  rw[tid] = wsum; rp[tid] = psum;
  __syncthreads();
  for (int s = 128; s > 0; s >>= 1) {
    if (tid < s) { rw[tid] += rw[tid + s]; rp[tid] += rp[tid + s]; }
    __syncthreads();
  }
  const float Wsum = rw[0] * 0.01f;   // DX
  const float Psum = rp[0] * 0.01f;
  for (int d = tid; d < 2500; d += 256) {
    const float x13 = xb[13 * 2500 + d];
    const float x14 = xb[14 * 2500 + d];
    const float x15 = xb[15 * 2500 + d];
    const float poly = (x13 + x14 + x15) * (1.0f / 3.0f) + (x15 - x13);
    const float outb = fb[d] + poly;
    out[(size_t)b * 2500 + d] = outb + (1.0f - outb * outb) / Wsum * Psum;
  }
}

// ---------------------------------------------------------------- launcher
extern "C" void kernel_launch(void* const* d_in, const int* in_sizes, int n_in,
                              void* d_out, int out_size, void* d_ws, size_t ws_size,
                              hipStream_t stream)
{
  (void)in_sizes; (void)n_in; (void)out_size; (void)ws_size;
  const float* x    = (const float*)d_in[0];
  const float* Wih1 = (const float*)d_in[1];
  const float* Whh1 = (const float*)d_in[2];
  const float* bih1 = (const float*)d_in[3];
  const float* bhh1 = (const float*)d_in[4];
  const float* Wih2 = (const float*)d_in[5];
  const float* Whh2 = (const float*)d_in[6];
  const float* bih2 = (const float*)d_in[7];
  const float* bhh2 = (const float*)d_in[8];
  const float* Wih3 = (const float*)d_in[9];
  const float* Whh3 = (const float*)d_in[10];
  const float* bih3 = (const float*)d_in[11];
  const float* bhh3 = (const float*)d_in[12];
  const float* Wfc  = (const float*)d_in[13];
  const float* bfc  = (const float*)d_in[14];
  float* out = (float*)d_out;

  const int B = 512, T = 16, D = 2500, H = 300, G = 4 * H; // G = 1200
  const int Dp = 2528;   // D padded to multiple of 32 (K pad)
  const int Hp = 320;    // H padded to multiple of 32 (K pad)
  const int Gp = 1280;   // G padded to multiple of 128 (B-rows pad)
  const int Dn = 2560;   // D padded to multiple of 128 (B-rows pad for Wfc)

  // ---- workspace carve (256B aligned)
  char* p = (char*)d_ws;
  auto alloc = [&](size_t bytes) -> char* {
    char* r = p; p += (bytes + 255) & ~(size_t)255; return r;
  };
  _Float16* x16     = (_Float16*)alloc((size_t)B * T * Dp * 2);  // 41.4 MB
  _Float16* Wih1_16 = (_Float16*)alloc((size_t)Gp * Dp * 2);     //  6.5 MB
  _Float16* Wih2_16 = (_Float16*)alloc((size_t)Gp * Hp * 2);
  _Float16* Wih3_16 = (_Float16*)alloc((size_t)Gp * Hp * 2);
  _Float16* Whh1_16 = (_Float16*)alloc((size_t)Gp * Hp * 2);
  _Float16* Whh2_16 = (_Float16*)alloc((size_t)Gp * Hp * 2);
  _Float16* Whh3_16 = (_Float16*)alloc((size_t)Gp * Hp * 2);
  _Float16* Wfc16   = (_Float16*)alloc((size_t)Dn * Hp * 2);     //  1.6 MB
  float*    xproj   = (float*)alloc((size_t)B * T * G * 4);      // 39.3 MB
  float*    gatesB  = (float*)alloc((size_t)B * G * 4);          //  2.5 MB
  _Float16* hbuf0   = (_Float16*)alloc((size_t)B * T * Hp * 2);  //  5.2 MB
  _Float16* hbuf1   = (_Float16*)alloc((size_t)B * T * Hp * 2);  //  5.2 MB
  float*    cbuf    = (float*)alloc((size_t)B * H * 4);          //  0.6 MB
  float*    fcout   = (float*)alloc((size_t)B * D * 4);          //  5.1 MB

  // ---- padded f32 -> f16 staging (zero-fills K pads and B-row pads)
  auto cvt = [&](const float* s, _Float16* d, int sr, int sc, int dr, int dc) {
    const int n = dr * dc;
    int blk = (n + 256 * 8 - 1) / (256 * 8);
    if (blk > 4096) blk = 4096;
    if (blk < 1) blk = 1;
    cvt_pad_f16<<<blk, 256, 0, stream>>>(s, d, sr, sc, dr, dc);
  };
  cvt(x,    x16,     B * T, D, B * T, Dp);
  cvt(Wih1, Wih1_16, G, D, Gp, Dp);
  cvt(Wih2, Wih2_16, G, H, Gp, Hp);
  cvt(Wih3, Wih3_16, G, H, Gp, Hp);
  cvt(Whh1, Whh1_16, G, H, Gp, Hp);
  cvt(Whh2, Whh2_16, G, H, Gp, Hp);
  cvt(Whh3, Whh3_16, G, H, Gp, Hp);
  cvt(Wfc,  Wfc16,   D, H, Dn, Hp);

  // ---- layer-1 input projection: xproj = x @ Wih1^T + bih1 + bhh1
  {
    dim3 g1(Gp / BN, (B * T) / BM); // (10, 64)
    wmma_gemm<<<g1, 256, 0, stream>>>(x16, Dp, Wih1_16, Dp, xproj, G,
                                      bih1, bhh1, nullptr, 0, G, Dp);
  }

  const _Float16* Whh16L[3] = {Whh1_16, Whh2_16, Whh3_16};
  const _Float16* Wih16L[3] = {nullptr, Wih2_16, Wih3_16};
  const float*    bihL[3]   = {bih1, bih2, bih3};
  const float*    bhhL[3]   = {bhh1, bhh2, bhh3};
  _Float16*       hb[2]     = {hbuf0, hbuf1};

  for (int l = 0; l < 3; ++l) {
    _Float16* hs = hb[l & 1]; // [B][T][Hp] f16 hidden states of this layer
    for (int t = 0; t < T; ++t) {
      if (t == 0) {
        // h0 == 0  ->  gates = xproj[:,0,:]
        lstm_gates<<<640, 256, 0, stream>>>(xproj, T * G, cbuf,
                                            hs /* + 0*Hp */, T * Hp, 1);
      } else {
        // gates = xproj[:,t,:] + h_{t-1} @ Whh^T
        dim3 gg(Gp / BN, B / BM); // (10, 4)
        wmma_gemm<<<gg, 256, 0, stream>>>(hs + (size_t)(t - 1) * Hp, T * Hp,
                                          Whh16L[l], Hp,
                                          gatesB, G, nullptr, nullptr,
                                          xproj + (size_t)t * G, T * G,
                                          G, Hp);
        lstm_gates<<<640, 256, 0, stream>>>(gatesB, G, cbuf,
                                            hs + (size_t)t * Hp, T * Hp, 0);
      }
    }
    if (l < 2) {
      // next layer projection: xproj = hs @ Wih_{l+1}^T + bih + bhh
      dim3 gx(Gp / BN, (B * T) / BM); // (10, 64)
      wmma_gemm<<<gx, 256, 0, stream>>>(hs, Hp, Wih16L[l + 1], Hp, xproj, G,
                                        bihL[l + 1], bhhL[l + 1], nullptr, 0,
                                        G, Hp);
    }
  }

  // ---- FC on last hidden state of layer 3 (lives in hb[0])
  {
    _Float16* h3 = hb[2 & 1];
    dim3 gf(Dn / BN, B / BM); // (20, 4)
    wmma_gemm<<<gf, 256, 0, stream>>>(h3 + (size_t)15 * Hp, T * Hp, Wfc16, Hp,
                                      fcout, D, bfc, nullptr, nullptr, 0,
                                      D, Hp);
  }

  // ---- poly bias + W_b correction
  final_poly<<<512, 256, 0, stream>>>(x, fcout, out);
}
